// Corr_Layer_21706764714774
// MI455X (gfx1250) — matrix operand order repaired and checked by reading, not compiled, not run on hardware
//
#include <hip/hip_runtime.h>
#include <hip/hip_bf16.h>
#include <cstdint>

// Problem constants (match reference)
#define BB   8
#define LL   4096
#define DD   1024
#define TOPK 16        // int(2 * ln(4096)) = 16

typedef __attribute__((ext_vector_type(16))) __bf16 v16bf;
typedef __attribute__((ext_vector_type(8)))  float  v8f;

// ---------- fp32 -> bf16 (round-to-nearest via +0x8000, 1-2 VALU ops) ----------
__device__ __forceinline__ unsigned short f2bf(float f) {
  return (unsigned short)((__float_as_uint(f) + 0x8000u) >> 16);
}
__device__ __forceinline__ unsigned int pk2(float a, float b) {
  const unsigned ua = __float_as_uint(a) + 0x8000u;
  const unsigned ub = __float_as_uint(b) + 0x8000u;
#if defined(__has_builtin) && __has_builtin(__builtin_amdgcn_perm)
  // dst = { ub[31:16], ua[31:16] }  (sel 0-3 -> S1=ua, 4-7 -> S0=ub)
  return __builtin_amdgcn_perm(ub, ua, 0x07060302u);
#else
  return (ua >> 16) | (ub & 0xFFFF0000u);
#endif
}

// ---------- CDNA5 async global->LDS copy (ASYNCcnt-tracked DMA) ----------
// LDS byte offset = low 32 bits of the generic pointer (ISA 10.2).
__device__ __forceinline__ void async_copy_b128(void* lds_dst, const void* gsrc) {
  unsigned ldsoff = (unsigned)(uintptr_t)lds_dst;
  asm volatile("global_load_async_to_lds_b128 %0, %1, off"
               :: "v"(ldsoff), "v"(gsrc) : "memory");
}
__device__ __forceinline__ void wait_async_all() {
  asm volatile("s_wait_asynccnt 0x0" ::: "memory");
}

// =====================================================================
// Pre-pass 1: elementwise fp32 -> bf16 (activations). 4 elems/thread.
// =====================================================================
__global__ void __launch_bounds__(256)
f32_to_bf16_kernel(const float* __restrict__ in, unsigned short* __restrict__ out) {
  const size_t i4 = ((size_t)blockIdx.x * 256 + threadIdx.x) * 4;
  const float4 v = *reinterpret_cast<const float4*>(in + i4);
  uint2 p; p.x = pk2(v.x, v.y); p.y = pk2(v.z, v.w);
  *reinterpret_cast<uint2*>(out + i4) = p;
}

// =====================================================================
// Pre-pass 2: W[K][N] fp32 -> Wt[N][K] bf16 (32x32 LDS tile transpose)
// =====================================================================
__global__ void __launch_bounds__(256)
transpose_w_bf16(const float* __restrict__ W, unsigned short* __restrict__ Wt) {
  __shared__ unsigned short tile[32][33];
  const int kt = blockIdx.y * 32;
  const int nt = blockIdx.x * 32;
  const int r  = threadIdx.x >> 5;   // 0..7
  const int c  = threadIdx.x & 31;   // 0..31
  for (int i = 0; i < 4; ++i) {
    const int rr = r + i * 8;
    tile[rr][c] = f2bf(W[(size_t)(kt + rr) * DD + nt + c]);
  }
  __syncthreads();
  for (int i = 0; i < 4; ++i) {
    const int rr = r + i * 8;
    Wt[(size_t)(nt + rr) * DD + kt + c] = tile[c][rr];
  }
}

// =====================================================================
// GEMM: C[M,N] = A[M,K] * Wt[N,K]^T + bias. A, Wt bf16; C fp32.
// M = 32768, N = K = 1024. Block tile 128x128x32, 256 threads (8 waves).
// Each wave (4x2 grid) owns a 32x64 tile = 2x4 fragments of 16x16.
// LDS is double-buffered and filled ONLY by async global->LDS DMA:
// tile kb+1 streams in while WMMAs consume tile kb.
// TRANS_OUT=true  -> store C transposed to series layout [B][N][L]
// TRANS_OUT=false -> store C row-major [M][N]
// =====================================================================
#define BM 128
#define BN 128
#define BK 32
#define LDSS 40   // LDS row stride in bf16 elems (32 + 8 pad)

template<bool TRANS_OUT>
__global__ void __launch_bounds__(256)
gemm_bf16_wmma(const unsigned short* __restrict__ A,   // [M][K] bf16
               const unsigned short* __restrict__ Wt,  // [N][K] bf16
               const float* __restrict__ bias, float* __restrict__ C) {
  __shared__ unsigned short sA[2][BM * LDSS];
  __shared__ unsigned short sB[2][BN * LDSS];

  const int tid  = threadIdx.x;
  const int lane = tid & 31;
  const int wid  = tid >> 5;
  const int wm   = wid & 3;    // wave row  (0..3) -> 32 rows each
  const int wn   = wid >> 2;   // wave col  (0..1) -> 64 cols each
  const int mBlock = blockIdx.y * BM;
  const int nBlock = blockIdx.x * BN;

  const int half = lane >> 4;  // ISA lane-half selector
  const int l16  = lane & 15;

  v8f acc[2][4];
  for (int i = 0; i < 2; ++i)
    for (int j = 0; j < 4; ++j)
      for (int e = 0; e < 8; ++e) acc[i][j][e] = 0.0f;

  // issue one 128x32 bf16 tile pair as 4 async b128 copies per thread
#define ISSUE_TILE(KB, BUF)                                                   \
  for (int t = 0; t < 2; ++t) {                                               \
    const int c   = tid + t * 256;   /* b128 chunk id, 0..511 */              \
    const int row = c >> 2;          /* 4 chunks per 32-elem row */           \
    const int sub = c & 3;                                                    \
    async_copy_b128(&sA[BUF][row * LDSS + sub * 8],                           \
                    A  + (size_t)(mBlock + row) * DD + (KB) * BK + sub * 8);  \
    async_copy_b128(&sB[BUF][row * LDSS + sub * 8],                           \
                    Wt + (size_t)(nBlock + row) * DD + (KB) * BK + sub * 8);  \
  }

  ISSUE_TILE(0, 0);

  for (int kb = 0; kb < DD / BK; ++kb) {
    const int cur = kb & 1;
    wait_async_all();                  // tile kb landed (this wave's copies)
    __syncthreads();                   // ..and everyone's; prev buf readers done
    if (kb + 1 < DD / BK) { ISSUE_TILE(kb + 1, cur ^ 1); }

    // ---- load fragments from LDS per ISA 7.12.2 bf16 layouts ----
    // A 16x32: lanes<16 hold K {0..7,16..23}, lanes>=16 hold {8..15,24..31}
    v16bf aF[2];
    for (int fm = 0; fm < 2; ++fm) {
      const int m = wm * 32 + fm * 16 + l16;
      union { uint4 u[2]; v16bf v; } cv;
      cv.u[0] = *reinterpret_cast<const uint4*>(&sA[cur][m * LDSS + half * 8]);
      cv.u[1] = *reinterpret_cast<const uint4*>(&sA[cur][m * LDSS + half * 8 + 16]);
      aF[fm] = cv.v;
    }
    // B 32x16: lanes<16 hold K {0..7,16..23} at col n=lane, lanes>=16 hold {8..15,24..31}
    v16bf bF[4];
    for (int fn = 0; fn < 4; ++fn) {
      const int n = wn * 64 + fn * 16 + l16;
      union { uint4 u[2]; v16bf v; } cv;
      cv.u[0] = *reinterpret_cast<const uint4*>(&sB[cur][n * LDSS + half * 8]);
      cv.u[1] = *reinterpret_cast<const uint4*>(&sB[cur][n * LDSS + half * 8 + 16]);
      bF[fn] = cv.v;
    }

    for (int fm = 0; fm < 2; ++fm)
      for (int fn = 0; fn < 4; ++fn)
        acc[fm][fn] = __builtin_amdgcn_wmma_f32_16x16x32_bf16(
            false, aF[fm], false, bF[fn], (short)0, acc[fm][fn], false, false);
  }
#undef ISSUE_TILE

  // ---- epilogue: bias + store ----
  // C fragment layout: VGPR r -> M = r + 8*(lane>=16), N = lane&15
  for (int fm = 0; fm < 2; ++fm) {
    for (int fn = 0; fn < 4; ++fn) {
      const int mg0 = mBlock + wm * 32 + fm * 16 + half * 8; // +r, r=0..7
      const int ng  = nBlock + wn * 64 + fn * 16 + l16;
      const float bv = bias[ng];
      if (TRANS_OUT) {
        // series layout: C[((b*DD + ng) * LL) + l], l contiguous in r
        const int b  = mg0 >> 12;        // mg0 / 4096
        const int l0 = mg0 & (LL - 1);
        float* p = C + ((size_t)b * DD + ng) * LL + l0;
        float4 s0, s1;
        s0.x = acc[fm][fn][0] + bv; s0.y = acc[fm][fn][1] + bv;
        s0.z = acc[fm][fn][2] + bv; s0.w = acc[fm][fn][3] + bv;
        s1.x = acc[fm][fn][4] + bv; s1.y = acc[fm][fn][5] + bv;
        s1.z = acc[fm][fn][6] + bv; s1.w = acc[fm][fn][7] + bv;
        *reinterpret_cast<float4*>(p)     = s0;
        *reinterpret_cast<float4*>(p + 4) = s1;
      } else {
        for (int r = 0; r < 8; ++r)
          C[(size_t)(mg0 + r) * DD + ng] = acc[fm][fn][r] + bv;
      }
    }
  }
}

// =====================================================================
// In-LDS radix-2 FFT, n=4096, 1024 threads. dir=+1 forward, -1 inverse
// (inverse leaves result un-normalized; 1/N folded into softmax input).
// =====================================================================
__device__ void fft4096(float* re, float* im, const float dir, const int tid) {
  __syncthreads();
  // bit-reversal permutation (12 bits)
  for (int i = tid; i < LL; i += 1024) {
    int j = (int)(__brev((unsigned)i) >> 20);
    if (j > i) {
      float tr = re[i]; re[i] = re[j]; re[j] = tr;
      float ti = im[i]; im[i] = im[j]; im[j] = ti;
    }
  }
  __syncthreads();
  for (int st = 1; st <= 12; ++st) {
    const int   halfn = 1 << (st - 1);
    const float ang0  = dir * (-6.28318530717958647692f) / (float)(1 << st);
    for (int bfy = tid; bfy < (LL / 2); bfy += 1024) {
      const int pos = bfy & (halfn - 1);
      const int grp = bfy >> (st - 1);
      const int i0  = (grp << st) + pos;
      const int i1  = i0 + halfn;
      float s, c;
      __sincosf(ang0 * (float)pos, &s, &c);
      const float xr = re[i1], xi = im[i1];
      const float tr = xr * c - xi * s;
      const float ti = xr * s + xi * c;
      re[i1] = re[i0] - tr; im[i1] = im[i0] - ti;
      re[i0] += tr;         im[i0] += ti;
    }
    __syncthreads();
  }
}

// =====================================================================
// One workgroup per series (b, h*128+d): FFT autocorrelation, top-16
// delays, softmax, delay aggregation of V. Series loads via CDNA5 async
// global->LDS DMA. Writes agg as bf16 in [B][L][D] (A of output GEMM).
// =====================================================================
__global__ void __launch_bounds__(1024)
autocorr_topk_agg(const float* __restrict__ Qp, const float* __restrict__ Kp,
                  const float* __restrict__ Vp, unsigned short* __restrict__ AGG) {
  __shared__ float qre[LL], qim[LL], kre[LL], kim[LL];  // 64 KB
  const int tid = threadIdx.x;
  const int s   = blockIdx.x;                // series id = b*1024 + col
  const size_t base = (size_t)s * LL;

  // ---- async-stage q and k series into LDS; zero imag parts ----
  {
    const int i4 = tid * 4;                  // one b128 per thread
    async_copy_b128(&qre[i4], Qp + base + i4);
    async_copy_b128(&kre[i4], Kp + base + i4);
    for (int e = 0; e < 4; ++e) { qim[i4 + e] = 0.f; kim[i4 + e] = 0.f; }
    wait_async_all();
  }
  __syncthreads();

  fft4096(qre, qim, 1.0f, tid);   // FFT(q)
  fft4096(kre, kim, 1.0f, tid);   // FFT(k)

  // P = Q * conj(K)  (each thread touches only its own indices)
  for (int i = tid; i < LL; i += 1024) {
    const float qr = qre[i], qi = qim[i], kr = kre[i], ki = kim[i];
    qre[i] = qr * kr + qi * ki;
    qim[i] = qi * kr - qr * ki;
  }
  __syncthreads();                // all kre readers done

  // ---- async-stage V into dead kre; overlaps inverse FFT + top-k ----
  async_copy_b128(&kre[tid * 4], Vp + base + tid * 4);

  // inverse FFT -> corr (real part in qre, scaled by N; fixed in softmax)
  fft4096(qre, qim, -1.0f, tid);

  // top-16 via iterative block argmax; scratch in dead kim
  float* sval = kim;
  int*   sidx = (int*)(kim + 1024);
  float* topw = kim + 2048;
  int*   topd = (int*)(kim + 2064);
  for (int r = 0; r < TOPK; ++r) {
    float bv = -3.0e38f; int bi = 0;
    for (int j = 0; j < 4; ++j) {
      const int l = tid + (j << 10);
      const float v = qre[l];
      if (v > bv) { bv = v; bi = l; }
    }
    sval[tid] = bv; sidx[tid] = bi;
    __syncthreads();
    for (int off = 512; off > 0; off >>= 1) {
      if (tid < off) {
        if (sval[tid + off] > sval[tid]) {
          sval[tid] = sval[tid + off];
          sidx[tid] = sidx[tid + off];
        }
      }
      __syncthreads();
    }
    if (tid == 0) {
      topw[r] = sval[0]; topd[r] = sidx[0];
      qre[sidx[0]] = -3.0e38f;          // knock out winner
    }
    __syncthreads();
  }

  // softmax over top-16 (every thread, registers) -- corr scaled by 1/N here
  float w[TOPK]; int dly[TOPK];
  float mx = -3.0e38f;
  for (int j = 0; j < TOPK; ++j) {
    const float t = topw[j] * (1.0f / (float)LL);
    w[j] = t; dly[j] = topd[j];
    if (t > mx) mx = t;
  }
  float ssum = 0.f;
  for (int j = 0; j < TOPK; ++j) { w[j] = __expf(w[j] - mx); ssum += w[j]; }
  const float inv = 1.0f / ssum;

  // V async copy must be complete and visible to all waves
  wait_async_all();
  __syncthreads();

  // delay aggregation: agg[l] = sum_j w[j] * v[(l + d_j) mod L]; store bf16
  const int b   = s >> 10;
  const int col = s & 1023;
  unsigned short* outp = AGG + (size_t)b * LL * DD + col;
  for (int jj = 0; jj < 4; ++jj) {
    const int l = tid + (jj << 10);
    float a = 0.f;
    for (int j = 0; j < TOPK; ++j) a += w[j] * kre[(l + dly[j]) & (LL - 1)];
    outp[(size_t)l * DD] = f2bf(a * inv);   // [B][L][D] bf16; L2-resident
  }
}

// =====================================================================
// Host-side launcher
// =====================================================================
extern "C" void kernel_launch(void* const* d_in, const int* in_sizes, int n_in,
                              void* d_out, int out_size, void* d_ws, size_t ws_size,
                              hipStream_t stream) {
  const float* q  = (const float*)d_in[0];
  const float* k  = (const float*)d_in[1];
  const float* v  = (const float*)d_in[2];
  const float* Wq = (const float*)d_in[3];
  const float* bq = (const float*)d_in[4];
  const float* Wk = (const float*)d_in[5];
  const float* bk = (const float*)d_in[6];
  const float* Wv = (const float*)d_in[7];
  const float* bv = (const float*)d_in[8];
  const float* Wo = (const float*)d_in[9];
  const float* bo = (const float*)d_in[10];
  float* out = (float*)d_out;

  // workspace: Qp|Kp|Vp fp32 (series layout) | Xq|Xk|Xv bf16 | Wt* bf16.
  // AGG (bf16, [B][L][D]) aliases Xq, which is dead once GEMM #1 finishes.
  const size_t NS = (size_t)BB * DD * LL;           // 33,554,432
  float* Qp = (float*)d_ws;
  float* Kp = Qp + NS;
  float* Vp = Kp + NS;
  unsigned short* Xq  = (unsigned short*)(Vp + NS);
  unsigned short* Xk  = Xq + NS;
  unsigned short* Xv  = Xk + NS;
  unsigned short* Wtq = Xv + NS;
  unsigned short* Wtk = Wtq + (size_t)DD * DD;
  unsigned short* Wtv = Wtk + (size_t)DD * DD;
  unsigned short* Wto = Wtv + (size_t)DD * DD;
  unsigned short* AGG = Xq;                          // alias

  // pre-pass: activations -> bf16, weights -> transposed bf16
  const int cvtBlocks = (int)(NS / 1024);            // 4 elems/thread
  f32_to_bf16_kernel<<<cvtBlocks, 256, 0, stream>>>(q, Xq);
  f32_to_bf16_kernel<<<cvtBlocks, 256, 0, stream>>>(k, Xk);
  f32_to_bf16_kernel<<<cvtBlocks, 256, 0, stream>>>(v, Xv);
  dim3 tgrid(DD / 32, DD / 32);
  transpose_w_bf16<<<tgrid, 256, 0, stream>>>(Wq, Wtq);
  transpose_w_bf16<<<tgrid, 256, 0, stream>>>(Wk, Wtk);
  transpose_w_bf16<<<tgrid, 256, 0, stream>>>(Wv, Wtv);
  transpose_w_bf16<<<tgrid, 256, 0, stream>>>(Wo, Wto);

  dim3 grid(DD / BN, (BB * LL) / BM);                // (8, 256)
  gemm_bf16_wmma<true ><<<grid, 256, 0, stream>>>(Xq, Wtq, bq, Qp);
  gemm_bf16_wmma<true ><<<grid, 256, 0, stream>>>(Xk, Wtk, bk, Kp);
  gemm_bf16_wmma<true ><<<grid, 256, 0, stream>>>(Xv, Wtv, bv, Vp);
  autocorr_topk_agg<<<BB * DD, 1024, 0, stream>>>(Qp, Kp, Vp, AGG);
  gemm_bf16_wmma<false><<<grid, 256, 0, stream>>>(AGG, Wto, bo, out);
}
